// AM_EGCL_77266461655561
// MI455X (gfx1250) — matrix operand order repaired and checked by reading, not compiled, not run on hardware
//
#include <hip/hip_runtime.h>
#include <hip/hip_bf16.h>

// EGCL layer for MI455X (gfx1250). Compute-bound: ~56 GFLOP of edge-MLP GEMMs,
// ~45MB working set (L2-resident). Strategy: 16-edge tiles per wave32,
// f16 activations/weights with f32 accumulation via v_wmma_f32_16x16x32_f16.

#define NN 50000
#define NE 400000
#define DF 128
#define EPSv 0.001f
#define BN_EPSv 1e-5f

typedef __attribute__((ext_vector_type(16))) _Float16 v16h;
typedef __attribute__((ext_vector_type(8)))  float    v8f;

struct __align__(16) U128 { unsigned int a, b, c, d; };
union FragU { v16h h; U128 u[2]; };

__device__ __forceinline__ float silu_f(float x) { return x / (1.0f + __expf(-x)); }

// intra-wave LDS producer->consumer ordering (no block barrier: waves are independent)
__device__ __forceinline__ void wsync() {
    __builtin_amdgcn_wave_barrier();
    asm volatile("s_wait_dscnt 0" ::: "memory");
    __builtin_amdgcn_wave_barrier();
}

__device__ __forceinline__ v8f wmma32(v16h a, v16h b, v8f c) {
    return __builtin_amdgcn_wmma_f32_16x16x32_f16(false, a, false, b, (short)0, c, false, false);
}

// A-fragment (16x32 f16) from row-major LDS [16][strideH] halves.
// lane<16: M=lane, K = kb..kb+7 and kb+16..kb+23 with kb=0; lane>=16: kb=8.
__device__ __forceinline__ v16h ldA(const _Float16* A, int strideH, int kt, int lane) {
    int m  = lane & 15;
    int kb = (lane & 16) ? 8 : 0;
    const _Float16* p = A + m * strideH + kt * 32 + kb;
    FragU f;
    f.u[0] = *reinterpret_cast<const U128*>(p);
    f.u[1] = *reinterpret_cast<const U128*>(p + 16);
    return f.h;
}

// B-fragment (32x16 f16): 32B contiguous per lane (pre-swizzled).
__device__ __forceinline__ v16h ldB(const _Float16* frag, int lane) {
    const U128* p = reinterpret_cast<const U128*>(frag + lane * 16);
    FragU f;
    f.u[0] = p[0];
    f.u[1] = p[1];
    return f.h;
}

// Pre-swizzle a KxN f32 weight matrix (row-major) into f16 B-fragments.
// frag = nt*Kt + kt; within frag: lane holds col nt*16+(lane&15),
// K values kt*32 + (lane>>4)*8 + (j&7) + (j>>3)*16, j=0..15.
__device__ void build_frags(_Float16* dst, const float* W, int K, int Kt, int N, int Nt,
                            int tid, int nth) {
    int total = Kt * Nt * 512;
    for (int e = tid; e < total; e += nth) {
        int frag = e >> 9;
        int lane = (e >> 4) & 31;
        int j    = e & 15;
        int nt = frag / Kt, kt = frag - nt * Kt;
        int n = nt * 16 + (lane & 15);
        int k = kt * 32 + ((lane >> 4) << 3) + (j & 7) + ((j >> 3) << 4);
        float v = (k < K && n < N) ? W[k * N + n] : 0.0f;
        dst[e] = (_Float16)v;
    }
}

// ---------------- Kernel A: global max pair distance ----------------
__global__ void k_gmax(const float* __restrict__ coords, const int* __restrict__ el,
                       unsigned int* __restrict__ gmax) {
    int e = blockIdx.x * blockDim.x + threadIdx.x;
    float m2 = 0.0f;
    if (e < NE) {
        int s = el[e * 3 + 0], t = el[e * 3 + 1];
        float ct[24], cs[24];
#pragma unroll
        for (int i = 0; i < 24; ++i) { ct[i] = coords[t * 24 + i]; cs[i] = coords[s * 24 + i]; }
#pragma unroll
        for (int ci = 0; ci < 8; ++ci)
#pragma unroll
            for (int cj = 0; cj < 8; ++cj) {
                float dx = ct[ci * 3 + 0] - cs[cj * 3 + 0];
                float dy = ct[ci * 3 + 1] - cs[cj * 3 + 1];
                float dz = ct[ci * 3 + 2] - cs[cj * 3 + 2];
                m2 = fmaxf(m2, dx * dx + dy * dy + dz * dz);
            }
    }
#pragma unroll
    for (int off = 16; off; off >>= 1) m2 = fmaxf(m2, __shfl_xor(m2, off, 32));
    if ((threadIdx.x & 31) == 0) atomicMax(gmax, __float_as_uint(sqrtf(m2)));
}

// ---------------- Kernel B: per-edge tile pipeline ----------------
// LDS head: w1f(73728B) w2f(32768) c1f(32768) c2f(4096) b1/b2/cb1(1536) rlw(2048) rlb(32)
// per-wave (17600B): Abuf 9216, Bbuf 4096, cdiff 1536, ef 512, tga 64, ewa 64, wpool 64, stg 2048
__global__ void __launch_bounds__(256) k_edges(
    const float* __restrict__ h, const float* __restrict__ coords,
    const float* __restrict__ cattr, const float* __restrict__ chw,
    const float* __restrict__ ewt, const int* __restrict__ el,
    const float* __restrict__ rl_w, const float* __restrict__ rl_b,
    const float* __restrict__ m_w1, const float* __restrict__ m_b1,
    const float* __restrict__ m_w2, const float* __restrict__ m_b2,
    const float* __restrict__ c_w1, const float* __restrict__ c_b1,
    const float* __restrict__ c_w2, const float* __restrict__ w_r,
    const float* __restrict__ gmaxp,
    float* __restrict__ nagg, float* __restrict__ csumG, float* __restrict__ cnt) {
    extern __shared__ char smem[];
    _Float16* w1f = (_Float16*)smem;                   // 9*8*512
    _Float16* w2f = w1f + 36864;                       // 4*8*512
    _Float16* c1f = w2f + 16384;                       // 4*8*512
    _Float16* c2f = c1f + 16384;                       // 4*1*512
    float* b1  = (float*)(smem + 143360);
    float* b2  = b1 + 128;
    float* cb1 = b2 + 128;
    float* rlw = cb1 + 128;                            // 512
    float* rlb = rlw + 512;                            // 8
    char* pw = smem + 146976;

    int tid = threadIdx.x, lane = tid & 31, wave = tid >> 5;
    char* mb = pw + wave * 17600;
    _Float16* Abuf = (_Float16*)mb;                    // 16x288 halves
    _Float16* Bbuf = (_Float16*)(mb + 9216);           // 16x128 halves
    float* cdiff  = (float*)(mb + 13312);              // 16x24
    float* ef     = (float*)(mb + 14848);              // 16x8
    int*   tga    = (int*)(mb + 15360);                // 16
    float* ewa    = (float*)(mb + 15424);              // 16
    float* wpool  = (float*)(mb + 15488);              // 16
    float* stg    = (float*)(mb + 15552);              // 512 floats

    build_frags(w1f, m_w1, 264, 9, 128, 8, tid, 256);
    build_frags(w2f, m_w2, 128, 4, 128, 8, tid, 256);
    build_frags(c1f, c_w1, 128, 4, 128, 8, tid, 256);
    build_frags(c2f, c_w2, 128, 4,   8, 1, tid, 256);
    if (tid < 128) { b1[tid] = m_b1[tid]; b2[tid] = m_b2[tid]; cb1[tid] = c_b1[tid]; }
    for (int i = tid; i < 512; i += 256) rlw[i] = rl_w[i];
    if (tid < 8) rlb[tid] = rl_b[tid];
    __syncthreads();

    const float gscale = 1.0f / (gmaxp[0] + EPSv);
    const float w_r0 = w_r[0];
    float* ct = stg;        float* cs = stg + 24;
    float* cwt = stg + 48;  float* cws = stg + 56;
    float* aT = stg + 64;   float* aS = stg + 128;
    float* Mm = stg + 192;  float* Tt = stg + 256;
    float* Rr = stg + 320;  float* ps = stg + 384;
    const int rbase = (lane & 16) ? 8 : 0;

    for (int tile = blockIdx.x * 8 + wave; tile < NE / 16; tile += gridDim.x * 8) {
        int base = tile * 16;
        // ---- preprocessing: build A = [h_t | h_s | radial | 0pad] (16 x 288 f16) ----
        for (int eL = 0; eL < 16; ++eL) {
            int eg = base + eL;
            int s = el[eg * 3 + 0], t = el[eg * 3 + 1];
            if (lane < 24) { ct[lane] = coords[t * 24 + lane]; cs[lane] = coords[s * 24 + lane]; }
            if (lane < 8)  { cwt[lane] = chw[t * 8 + lane];    cws[lane] = chw[s * 8 + lane]; }
            aT[lane] = cattr[t * 64 + lane]; aT[lane + 32] = cattr[t * 64 + lane + 32];
            aS[lane] = cattr[s * 64 + lane]; aS[lane + 32] = cattr[s * 64 + lane + 32];
            wsync();
#pragma unroll
            for (int q = 0; q < 2; ++q) {  // 64 scaled pair distances
                int idx = lane + q * 32; int ci = idx >> 3, cj = idx & 7;
                float dx = ct[ci * 3 + 0] - cs[cj * 3 + 0];
                float dy = ct[ci * 3 + 1] - cs[cj * 3 + 1];
                float dz = ct[ci * 3 + 2] - cs[cj * 3 + 2];
                Mm[idx] = sqrtf(dx * dx + dy * dy + dz * dz) * gscale * cwt[ci] * cws[cj];
            }
            wsync();
#pragma unroll
            for (int q = 0; q < 2; ++q) {  // T = M @ aS
                int idx = lane + q * 32; int c = idx >> 3, b = idx & 7;
                float acc = 0.0f;
#pragma unroll
                for (int d = 0; d < 8; ++d) acc += Mm[c * 8 + d] * aS[d * 8 + b];
                Tt[idx] = acc;
            }
            wsync();
            float ss = 0.0f;
#pragma unroll
            for (int q = 0; q < 2; ++q) {  // radial = aT^T @ T
                int idx = lane + q * 32; int a = idx >> 3, b = idx & 7;
                float acc = 0.0f;
#pragma unroll
                for (int c = 0; c < 8; ++c) acc += aT[c * 8 + a] * Tt[c * 8 + b];
                Rr[idx] = acc; ss += acc * acc;
            }
#pragma unroll
            for (int off = 16; off; off >>= 1) ss += __shfl_xor(ss, off, 32);
            float rinv = 1.0f / (sqrtf(ss) + EPSv);
            wsync();
            if (lane < 8) {  // radial-layer 64->8
                float acc = rlb[lane];
                for (int i = 0; i < 64; ++i) acc += Rr[i] * rinv * rlw[i * 8 + lane];
                Abuf[eL * 288 + 256 + lane] = (_Float16)acc;
            } else {
                Abuf[eL * 288 + 256 + lane] = (_Float16)0.0f;  // K pad 264..287
            }
#pragma unroll
            for (int r = 0; r < 4; ++r) {
                int col = lane + r * 32;
                Abuf[eL * 288 + col]       = (_Float16)h[t * 128 + col];
                Abuf[eL * 288 + 128 + col] = (_Float16)h[s * 128 + col];
            }
            float csum_s = 0.0f, csum_t = 0.0f;
#pragma unroll
            for (int c = 0; c < 8; ++c) {
                csum_s += (cws[c] != 0.0f) ? 1.0f : 0.0f;
                csum_t += (cwt[c] != 0.0f) ? 1.0f : 0.0f;
            }
            if (lane < 3) {
                float acc = 0.0f;
#pragma unroll
                for (int c = 0; c < 8; ++c) acc += cs[c * 3 + lane] * ((cws[c] != 0.0f) ? 1.0f : 0.0f);
                ps[lane] = acc / (csum_s + EPSv);
            }
            wsync();
            if (lane < 24) cdiff[eL * 24 + lane] = ct[lane] - ps[lane % 3];
            if (lane == 0) {
                tga[eL] = t; ewa[eL] = ewt[eg];
                int tsv = (int)csum_t - 1; tsv = tsv < 0 ? 0 : (tsv > 7 ? 7 : tsv);
                wpool[eL] = (float)(8 - tsv);
            }
            wsync();
        }
        // ---- L1: x(288) @ m_w1 -> silu -> Bbuf ----
        for (int nt = 0; nt < 8; ++nt) {
            int col = nt * 16 + (lane & 15);
            v8f acc; float bb = b1[col];
#pragma unroll
            for (int i = 0; i < 8; ++i) acc[i] = bb;
#pragma unroll
            for (int kt = 0; kt < 9; ++kt)
                acc = wmma32(ldA(Abuf, 288, kt, lane), ldB(w1f + (nt * 9 + kt) * 512, lane), acc);
#pragma unroll
            for (int i = 0; i < 8; ++i) Bbuf[(rbase + i) * 128 + col] = (_Float16)silu_f(acc[i]);
        }
        wsync();
        // ---- L2: -> m; scatter node_agg atomics; m(f16) -> Abuf[:, :128] ----
        for (int nt = 0; nt < 8; ++nt) {
            int col = nt * 16 + (lane & 15);
            v8f acc; float bb = b2[col];
#pragma unroll
            for (int i = 0; i < 8; ++i) acc[i] = bb;
#pragma unroll
            for (int kt = 0; kt < 4; ++kt)
                acc = wmma32(ldA(Bbuf, 128, kt, lane), ldB(w2f + (nt * 4 + kt) * 512, lane), acc);
#pragma unroll
            for (int i = 0; i < 8; ++i) {
                int row = rbase + i;
                float v = silu_f(acc[i]);
                Abuf[row * 288 + col] = (_Float16)v;
                atomicAdd(&nagg[tga[row] * 128 + col], v * ewa[row]);
            }
        }
        wsync();
        // ---- L3: silu(m @ c_w1 + c_b1) -> Bbuf ----
        for (int nt = 0; nt < 8; ++nt) {
            int col = nt * 16 + (lane & 15);
            v8f acc; float bb = cb1[col];
#pragma unroll
            for (int i = 0; i < 8; ++i) acc[i] = bb;
#pragma unroll
            for (int kt = 0; kt < 4; ++kt)
                acc = wmma32(ldA(Abuf, 288, kt, lane), ldB(c1f + (nt * 4 + kt) * 512, lane), acc);
#pragma unroll
            for (int i = 0; i < 8; ++i) Bbuf[(rbase + i) * 128 + col] = (_Float16)silu_f(acc[i]);
        }
        wsync();
        // ---- L4: edge_feat = (.) @ c_w2 (N=8 in one padded N-tile) ----
        {
            v8f acc;
#pragma unroll
            for (int i = 0; i < 8; ++i) acc[i] = 0.0f;
#pragma unroll
            for (int kt = 0; kt < 4; ++kt)
                acc = wmma32(ldA(Bbuf, 128, kt, lane), ldB(c2f + kt * 512, lane), acc);
            int c0 = lane & 15;
            if (c0 < 8) {
#pragma unroll
                for (int i = 0; i < 8; ++i) ef[(rbase + i) * 8 + c0] = acc[i];
            }
        }
        wsync();
        // ---- pool-matrix + coord_message scatter ----
#pragma unroll
        for (int q = 0; q < 4; ++q) {
            int tsk = lane + q * 32; int e2 = tsk >> 3, ch = tsk & 7;
            float w = wpool[e2];
            int hi = ch + (int)w; hi = hi > 8 ? 8 : hi;
            float acc = 0.0f;
            for (int c = ch; c < hi; ++c) acc += ef[e2 * 8 + c];
            float coeff = (acc / w) * w_r0;
            int tn = tga[e2];
#pragma unroll
            for (int d = 0; d < 3; ++d)
                atomicAdd(&csumG[tn * 24 + ch * 3 + d], cdiff[e2 * 24 + ch * 3 + d] * coeff);
        }
        if (lane < 16) atomicAdd(&cnt[tga[lane]], 1.0f);
        wsync();
    }
}

// ---------------- Kernel C: node update (WMMA) + coord epilogue ----------------
__global__ void __launch_bounds__(256) k_nodes(
    const float* __restrict__ h, const float* __restrict__ coords,
    const float* __restrict__ het_w, const float* __restrict__ het_b,
    const float* __restrict__ gamma, const float* __restrict__ beta,
    const float* __restrict__ nagg, const float* __restrict__ csumG,
    const float* __restrict__ cnt,
    float* __restrict__ nodeOut, float* __restrict__ coordOut) {
    extern __shared__ char smem[];
    _Float16* hf = (_Float16*)smem;                 // 4*8*512 halves
    float* hb = (float*)(smem + 32768);
    float* gm = hb + 128; float* bt = gm + 128;
    char* pw = smem + 32768 + 1536;
    int tid = threadIdx.x, lane = tid & 31, wave = tid >> 5;
    _Float16* Abuf = (_Float16*)(pw + wave * 4096); // 16x128 halves
    build_frags(hf, het_w, 128, 4, 128, 8, tid, 256);
    if (tid < 128) { hb[tid] = het_b[tid]; gm[tid] = gamma[tid]; bt[tid] = beta[tid]; }
    __syncthreads();
    const float invs = rsqrtf(1.0f + BN_EPSv);
    const int rbase = (lane & 16) ? 8 : 0;

    for (int tile = blockIdx.x * 8 + wave; tile < NN / 16; tile += gridDim.x * 8) {
        int base = tile * 16;
        for (int idx = lane; idx < 2048; idx += 32) {
            int r = idx >> 7, c = idx & 127;
            Abuf[idx] = (_Float16)nagg[(base + r) * 128 + c];
        }
        wsync();
        for (int nt = 0; nt < 8; ++nt) {
            int col = nt * 16 + (lane & 15);
            v8f acc; float bb = hb[col];
#pragma unroll
            for (int i = 0; i < 8; ++i) acc[i] = bb;
#pragma unroll
            for (int kt = 0; kt < 4; ++kt)
                acc = wmma32(ldA(Abuf, 128, kt, lane), ldB(hf + (nt * 4 + kt) * 512, lane), acc);
#pragma unroll
            for (int i = 0; i < 8; ++i) {
                int n = base + rbase + i;
                float v = silu_f(acc[i] * invs * gm[col] + bt[col]);
                nodeOut[n * 128 + col] = h[n * 128 + col] + v;
            }
        }
        for (int idx = lane; idx < 384; idx += 32) {
            int r = idx / 24, j = idx - r * 24; int n = base + r;
            float c = cnt[n]; c = c < 1.0f ? 1.0f : c;
            coordOut[n * 24 + j] = coords[n * 24 + j] + csumG[n * 24 + j] / c;
        }
        wsync();
    }
}

extern "C" void kernel_launch(void* const* d_in, const int* in_sizes, int n_in,
                              void* d_out, int out_size, void* d_ws, size_t ws_size,
                              hipStream_t stream) {
    const float* h      = (const float*)d_in[0];
    const float* coords = (const float*)d_in[1];
    const float* cattr  = (const float*)d_in[2];
    const float* chw    = (const float*)d_in[3];
    const float* ewt    = (const float*)d_in[4];
    const int*   el     = (const int*)  d_in[5];
    const float* rl_w   = (const float*)d_in[6];
    const float* rl_b   = (const float*)d_in[7];
    const float* m_w1   = (const float*)d_in[8];
    const float* m_b1   = (const float*)d_in[9];
    const float* m_w2   = (const float*)d_in[10];
    const float* m_b2   = (const float*)d_in[11];
    const float* c_w1   = (const float*)d_in[12];
    const float* c_b1   = (const float*)d_in[13];
    const float* c_w2   = (const float*)d_in[14];
    const float* het_w  = (const float*)d_in[15];
    const float* het_b  = (const float*)d_in[16];
    const float* gmm    = (const float*)d_in[17];
    const float* btt    = (const float*)d_in[18];
    const float* w_r    = (const float*)d_in[19];

    float* out = (float*)d_out;
    float* nagg  = out;                         // N*128 accumulator, finalized in-place
    float* csumG = out + (size_t)NN * 128;      // N*24 accumulator, finalized in-place
    unsigned int* gmax_u = (unsigned int*)d_ws; // 4B at offset 0
    float* cnt = (float*)((char*)d_ws + 256);   // N floats

    hipMemsetAsync(d_out, 0, (size_t)out_size * sizeof(float), stream);
    hipMemsetAsync(d_ws, 0, 256 + (size_t)NN * sizeof(float), stream);

    k_gmax<<<(NE + 255) / 256, 256, 0, stream>>>(coords, el, gmax_u);

    size_t ldsB = 146976 + 8 * 17600;  // 287,776 B (< 320KB WGP limit)
    k_edges<<<256, 256, ldsB, stream>>>(h, coords, cattr, chw, ewt, el, rl_w, rl_b,
                                        m_w1, m_b1, m_w2, m_b2, c_w1, c_b1, c_w2, w_r,
                                        (const float*)d_ws, nagg, csumG, cnt);

    size_t ldsC = 32768 + 1536 + 8 * 4096;  // 67,072 B
    k_nodes<<<160, 256, ldsC, stream>>>(h, coords, het_w, het_b, gmm, btt,
                                        nagg, csumG, cnt, nagg, csumG);
}
